// LightweightCNNMoE_66116726555019
// MI455X (gfx1250) — matrix operand order repaired and checked by reading, not compiled, not run on hardware
//
#include <hip/hip_runtime.h>

typedef __attribute__((ext_vector_type(16))) _Float16 v16h;
typedef __attribute__((ext_vector_type(8)))  _Float16 v8h;
typedef __attribute__((ext_vector_type(4)))  _Float16 v4h;
typedef __attribute__((ext_vector_type(8)))  float    v8f;

#define BN_EPS 1e-5f
#define WSTR 200   // wA row stride in halfs (400B: conflict-free b128 across 16 rows)

// conv(k=3,pad=1)+BN+ReLU as WMMA GEMM over transposed activations.
// K ordering: k = dl*Cin + ci  (Cin power of two -> shift/mask decode).
// Activations in LDS transposed: actT[row = l+1][c], row stride Cin+8 halfs,
// rows 0 and 61..67 are zero (conv halo). B[k][n] = actT[n+dl][ci] -> a
// 16-wide k-run is 16 contiguous halfs => 2x ds_load_b128 per fragment.
template<int Cout, int Cin, bool POOL>
__device__ __forceinline__ void conv_gemm(
    int tid,
    const float* __restrict__ wsrc,   // [Cout][Cin][3]
    const float* __restrict__ bsrc,   // [Cout]
    const float* __restrict__ bnsrc,  // [4][Cout] scale,bias,mean,var
    _Float16* wA,                     // LDS [64*WSTR]
    const _Float16* actIn,            // LDS transposed [68][Cin+8]
    _Float16* actOut,                 // LDS transposed [68][Cout+8] (POOL=false)
    int* poolMax,                     // LDS [Cout] (POOL=true, float bits)
    float* sinv, float* sbias)
{
    constexpr int Ktot   = 3 * Cin;
    constexpr int KLEN   = (Ktot > 192) ? 192 : Ktot;   // 384->2x192, else 1 chunk
    constexpr int NCHUNK = Ktot / KLEN;
    constexpr int L2C    = (Cin == 128) ? 7 : ((Cin == 64) ? 6 : 5);
    constexpr int ISTR   = Cin + 8;
    constexpr int OSTR   = Cout + 8;
    constexpr int MT = Cout / 16, NT = 4, MTNT = MT * NT;  // MTNT in {4,8,16}
    constexpr int TPW = (MTNT + 7) / 8;
    // staging decomposition: shifts/masks only (Cout, TPR powers of two)
    constexpr int TPR = 256 / Cout;     // threads per weight row: 4, 8, 16
    constexpr int KPT = KLEN / TPR;     // kk per thread: 48, 24, 6

    // Fold BN into per-channel scale + bias
    if (tid < Cout) {
        float sc = bnsrc[tid];
        float bi = bnsrc[Cout + tid];
        float mn = bnsrc[2 * Cout + tid];
        float vr = bnsrc[3 * Cout + tid];
        float inv = sc / sqrtf(vr + BN_EPS);
        sinv[tid]  = inv;
        sbias[tid] = bsrc[tid] * inv + bi - mn * inv;
    }
    __syncthreads();

    const int wave = tid >> 5;
    const int lane = tid & 31;
    const int hid  = lane >> 4;
    const int lr   = lane & 15;

    const int co_s  = tid / TPR;               // shift
    const int kk0_s = (tid & (TPR - 1)) * KPT; // mask + shift

    const v8f vzero = {};
    v8f acc[TPW];
#pragma unroll
    for (int i = 0; i < TPW; ++i) acc[i] = vzero;

#pragma unroll
    for (int kc = 0; kc < NCHUNK; ++kc) {
        __syncthreads();   // previous chunk of wA fully consumed
        // Stage BN-folded weights, reordered to k = dl*Cin + ci (no divisions)
        {
            const float invc = sinv[co_s];
            const float* wrow = wsrc + co_s * Ktot;
            _Float16* wdst = wA + co_s * WSTR + kk0_s;
            if constexpr ((KPT & 3) == 0) {
#pragma unroll
                for (int j = 0; j < KPT; j += 4) {
                    v4h h;
#pragma unroll
                    for (int u = 0; u < 4; ++u) {
                        int k  = kc * KLEN + kk0_s + j + u;
                        int ci = k & (Cin - 1);
                        int dl = k >> L2C;
                        h[u] = (_Float16)(wrow[ci * 3 + dl] * invc);
                    }
                    *(v4h*)(wdst + j) = h;   // one 8-byte ds store per 4 weights
                }
            } else {
#pragma unroll
                for (int j = 0; j < KPT; ++j) {
                    int k  = kc * KLEN + kk0_s + j;
                    int ci = k & (Cin - 1);
                    int dl = k >> L2C;
                    wdst[j] = (_Float16)(wrow[ci * 3 + dl] * invc);
                }
            }
        }
        __syncthreads();

#pragma unroll
        for (int ti = 0; ti < TPW; ++ti) {
            int t = (wave + ti * 8) & (MTNT - 1);  // idle waves duplicate a tile:
                                                   // EXEC stays all-1s through WMMA
            int mt = t / NT, nt = t % NT;
            const _Float16* ap = wA + (mt * 16 + lr) * WSTR;
            const int ncol = nt * 16 + lr;
#pragma unroll
            for (int kso = 0; kso < KLEN; kso += 32) {
                // A fragment (16x32 f16): two contiguous b128 LDS loads
                v8h a0 = *(const v8h*)(ap + kso + hid * 16);
                v8h a1 = *(const v8h*)(ap + kso + hid * 16 + 8);
                v16h af = __builtin_shufflevector(a0, a1,
                    0,1,2,3,4,5,6,7,8,9,10,11,12,13,14,15);
                // B fragment (32x16 f16): k = kc*KLEN + kso + hid*16 + i,
                // run of 16 contiguous ci at fixed dl -> two b128 LDS loads
                int k0  = kc * KLEN + kso + hid * 16;
                int dl  = k0 >> L2C;
                int ci0 = k0 & (Cin - 1);
                const _Float16* bp = actIn + (ncol + dl) * ISTR + ci0;
                v8h b0 = *(const v8h*)bp;
                v8h b1 = *(const v8h*)(bp + 8);
                v16h bf = __builtin_shufflevector(b0, b1,
                    0,1,2,3,4,5,6,7,8,9,10,11,12,13,14,15);
                acc[ti] = __builtin_amdgcn_wmma_f32_16x16x32_f16(
                    false, af, false, bf, (short)0, acc[ti], false, false);
            }
        }
    }

    // Epilogue: bias + ReLU, then pool or transposed vector store
#pragma unroll
    for (int ti = 0; ti < TPW; ++ti) {
        int t = (wave + ti * 8) & (MTNT - 1);
        int mt = t / NT, nt = t % NT;
        int n   = nt * 16 + lr;
        int co0 = mt * 16 + hid * 8;
        if (n < 60) {
            if (POOL) {
#pragma unroll
                for (int v = 0; v < 8; ++v) {
                    float val = fmaxf(acc[ti][v] + sbias[co0 + v], 0.0f);
                    atomicMax(poolMax + co0 + v, __float_as_int(val)); // relu>=0
                }
            } else {
                v8h o;
#pragma unroll
                for (int v = 0; v < 8; ++v)
                    o[v] = (_Float16)fmaxf(acc[ti][v] + sbias[co0 + v], 0.0f);
                *(v8h*)(actOut + (n + 1) * OSTR + co0) = o;  // one ds_store_b128
            }
        }
    }
    __syncthreads();
}

__global__ __launch_bounds__(256)
void moe_fused_kernel(
    const int*   __restrict__ x,          // [B,60]
    const float* __restrict__ emb,        // [64,128]
    const float* __restrict__ g_conv_w, const float* __restrict__ g_conv_b,
    const float* __restrict__ g_bn,
    const float* __restrict__ g_fc1_w,  const float* __restrict__ g_fc1_b,
    const float* __restrict__ g_fc2_w,  const float* __restrict__ g_fc2_b,
    const float* __restrict__ e_conv1_w, const float* __restrict__ e_conv1_b,
    const float* __restrict__ e_bn1,
    const float* __restrict__ e_conv2_w, const float* __restrict__ e_conv2_b,
    const float* __restrict__ e_bn2,
    const float* __restrict__ e_conv3_w, const float* __restrict__ e_conv3_b,
    const float* __restrict__ e_bn3,
    const float* __restrict__ fc1_w, const float* __restrict__ fc1_b,
    const float* __restrict__ fc2_w, const float* __restrict__ fc2_b,
    float* __restrict__ out)              // [B,2]
{
    __shared__ _Float16 wA[64 * WSTR];      // 25.6 KB weight chunk (BN-folded f16)
    __shared__ _Float16 act0T[68 * 136];    // 18.5 KB emb/conv1 input, [row][c=128]
    __shared__ _Float16 act1T[68 * 72];     //  9.8 KB conv1 out, [row][c=64]
    __shared__ _Float16 act2T[68 * 40];     //  5.4 KB conv2 out, [row][c=32]
    __shared__ float sinv[64], sbias[64];
    __shared__ int   smax[64];              // gating pooled (float bits)
    __shared__ int   pmax[16];              // expert pooled (float bits)
    __shared__ float sh32[32];
    __shared__ float slog[3];
    __shared__ int   sexpert;
    __shared__ float sh64[64];

    const int tid = threadIdx.x;
    const int b   = blockIdx.x;

    // Init pooled maxima (relu>=0 so 0-bits is identity) and halo rows {0,61..67}
    if (tid < 64) smax[tid] = 0;
    if (tid < 16) pmax[tid] = 0;
    if (tid < 136) {
#pragma unroll
        for (int r = 0; r < 8; ++r)
            act0T[(r ? 60 + r : 0) * 136 + tid] = (_Float16)0.f;
    }
    if (tid < 72) {
#pragma unroll
        for (int r = 0; r < 8; ++r)
            act1T[(r ? 60 + r : 0) * 72 + tid] = (_Float16)0.f;
    }
    if (tid < 40) {
#pragma unroll
        for (int r = 0; r < 8; ++r)
            act2T[(r ? 60 + r : 0) * 40 + tid] = (_Float16)0.f;
    }

    // Gather embedding -> act0T[l+1][c] in f16 (float4 loads, 8-byte v4h stores)
    const int* xb = x + (size_t)b * 60;
    for (int i = tid; i < 60 * 32; i += 256) {
        int l = i >> 5;
        int g = (i & 31) << 2;
        float4 f = *(const float4*)(emb + xb[l] * 128 + g);
        v4h h = { (_Float16)f.x, (_Float16)f.y, (_Float16)f.z, (_Float16)f.w };
        *(v4h*)(act0T + (l + 1) * 136 + g) = h;
    }
    __syncthreads();

    // ---- gating: conv+BN+ReLU -> maxpool over L ----
    conv_gemm<64, 128, true>(tid, g_conv_w, g_conv_b, g_bn,
                             wA, act0T, nullptr, smax, sinv, sbias);

    // ---- gating MLP + argmax (softmax monotonic -> argmax of logits) ----
    if (tid < 32) {
        float a = g_fc1_b[tid];
        for (int i = 0; i < 64; ++i)
            a += __int_as_float(smax[i]) * g_fc1_w[tid * 64 + i];
        sh32[tid] = fmaxf(a, 0.f);
    }
    __syncthreads();
    if (tid < 3) {
        float a = g_fc2_b[tid];
        for (int i = 0; i < 32; ++i)
            a += sh32[i] * g_fc2_w[tid * 32 + i];
        slog[tid] = a;
    }
    __syncthreads();
    if (tid == 0) {
        int be = 0; float bv = slog[0];
        if (slog[1] > bv) { bv = slog[1]; be = 1; }
        if (slog[2] > bv) { bv = slog[2]; be = 2; }
        sexpert = be;
    }
    __syncthreads();
    const int e = sexpert;

    // ---- selected expert only (== one-hot masked dense dispatch) ----
    conv_gemm<64, 128, false>(tid, e_conv1_w + e * 64 * 384, e_conv1_b + e * 64,
                              e_bn1 + e * 4 * 64, wA, act0T, act1T, nullptr, sinv, sbias);
    conv_gemm<32, 64, false>(tid, e_conv2_w + e * 32 * 192, e_conv2_b + e * 32,
                             e_bn2 + e * 4 * 32, wA, act1T, act2T, nullptr, sinv, sbias);
    conv_gemm<16, 32, true >(tid, e_conv3_w + e * 16 * 96,  e_conv3_b + e * 16,
                             e_bn3 + e * 4 * 16, wA, act2T, nullptr, pmax, sinv, sbias);

    // ---- head MLP ----
    if (tid < 64) {
        float a = fc1_b[tid];
        for (int i = 0; i < 16; ++i)
            a += __int_as_float(pmax[i]) * fc1_w[tid * 16 + i];
        sh64[tid] = fmaxf(a, 0.f);
    }
    __syncthreads();
    if (tid < 2) {
        float a = fc2_b[tid];
        for (int i = 0; i < 64; ++i)
            a += sh64[i] * fc2_w[tid * 64 + i];
        out[(size_t)b * 2 + tid] = a;
    }
}

extern "C" void kernel_launch(void* const* d_in, const int* in_sizes, int n_in,
                              void* d_out, int out_size, void* d_ws, size_t ws_size,
                              hipStream_t stream) {
    (void)n_in; (void)out_size; (void)d_ws; (void)ws_size;
    const int B = in_sizes[0] / 60;
    moe_fused_kernel<<<B, 256, 0, stream>>>(
        (const int*)  d_in[0],  (const float*)d_in[1],
        (const float*)d_in[2],  (const float*)d_in[3],  (const float*)d_in[4],
        (const float*)d_in[5],  (const float*)d_in[6],
        (const float*)d_in[7],  (const float*)d_in[8],
        (const float*)d_in[9],  (const float*)d_in[10], (const float*)d_in[11],
        (const float*)d_in[12], (const float*)d_in[13], (const float*)d_in[14],
        (const float*)d_in[15], (const float*)d_in[16], (const float*)d_in[17],
        (const float*)d_in[18], (const float*)d_in[19],
        (const float*)d_in[20], (const float*)d_in[21],
        (float*)d_out);
}
